// Attention_28192165330957
// MI455X (gfx1250) — matrix operand order repaired and checked by reading, not compiled, not run on hardware
//
#include <hip/hip_runtime.h>

typedef _Float16 f16;
typedef _Float16 v16h __attribute__((ext_vector_type(16)));
typedef float    v8f  __attribute__((ext_vector_type(8)));
typedef unsigned int v4u __attribute__((ext_vector_type(4)));
typedef int          v4i __attribute__((ext_vector_type(4)));

#define Bn     8
#define Nn     1024
#define DIMn   512
#define Hn     8
#define DHn    64
#define INNERn 512

#if defined(__has_builtin)
#if __has_builtin(__builtin_amdgcn_global_load_async_to_lds_b128)
#define HAS_ASYNC_LDS 1
#endif
#endif

union AF { v16h h; unsigned u[8]; };
union BF { v16h h; v4u q[2]; };
union PK { v4u q; f16 e[8]; };

// A-fragment (16x32 f16, M x K): lane holds row M=lane&15; K base = (lane<16?0:8)
// VGPR v covers K pair: v<4 -> base+2v ; v>=4 -> base+16+2(v-4)  (two contiguous 16B runs)
__device__ __forceinline__ v16h load_a_frag(const f16* __restrict__ row, int base) {
  AF a;
#pragma unroll
  for (int v = 0; v < 8; ++v) {
    int off = (v < 4) ? 2 * v : 8 + 2 * v;
    a.u[v] = *(const unsigned*)(row + base + off);
  }
  return a.h;
}

// B-fragment (32x16 f16, K x N): lane holds col N=lane&15; element s -> K = 8h+s (s<8), 16+8h+(s-8) (s>=8)
__device__ __forceinline__ v16h load_b_frag(const f16* __restrict__ p, int ld, int half) {
  v16h b;
#pragma unroll
  for (int s = 0; s < 16; ++s) {
    int k = (s < 8) ? (8 * half + s) : (16 + 8 * half + (s - 8));
    b[s] = p[k * ld];
  }
  return b;
}

__device__ __forceinline__ v8f wmma_f16(v16h a, v16h b, v8f c) {
  return __builtin_amdgcn_wmma_f32_16x16x32_f16(false, a, false, b, (short)0, c, false, false);
}

// 16-byte global -> LDS stage: async-DMA path on gfx1250, plain copy fallback.
__device__ __forceinline__ void stage16(const f16* g, f16* l) {
#ifdef HAS_ASYNC_LDS
  __builtin_amdgcn_global_load_async_to_lds_b128(
      (__attribute__((address_space(1))) v4i*)(f16*)g,
      (__attribute__((address_space(3))) v4i*)l, 0, 0);
#else
  *(v4u*)l = *(const v4u*)g;
#endif
}
__device__ __forceinline__ void stage_wait() {
#ifdef HAS_ASYNC_LDS
#if __has_builtin(__builtin_amdgcn_s_wait_asynccnt)
  __builtin_amdgcn_s_wait_asynccnt(0);
#else
  asm volatile("s_wait_asynccnt 0x0" ::: "memory");
#endif
#endif
}

// ---------------- LayerNorm: x f32 [8192,512] -> xn f16 ----------------
__global__ void ln_kernel(const float* __restrict__ x, const float* __restrict__ g,
                          const float* __restrict__ be, f16* __restrict__ xn) {
  __shared__ float red[256];
  int row = blockIdx.x, t = threadIdx.x;
  const float* xr = x + (size_t)row * DIMn;
  float a0 = xr[t], a1 = xr[t + 256];
  red[t] = a0 + a1;
  __syncthreads();
  for (int s = 128; s > 0; s >>= 1) { if (t < s) red[t] += red[t + s]; __syncthreads(); }
  float mu = red[0] * (1.0f / DIMn);
  __syncthreads();
  float d0 = a0 - mu, d1 = a1 - mu;
  red[t] = d0 * d0 + d1 * d1;
  __syncthreads();
  for (int s = 128; s > 0; s >>= 1) { if (t < s) red[t] += red[t + s]; __syncthreads(); }
  float rs = rsqrtf(red[0] * (1.0f / DIMn) + 1e-5f);
  f16* orow = xn + (size_t)row * DIMn;
  orow[t]       = (f16)(d0 * rs * g[t] + be[t]);
  orow[t + 256] = (f16)(d1 * rs * g[t + 256] + be[t + 256]);
}

// ---------------- f32 -> f16 convert ----------------
__global__ void cvt_kernel(const float* __restrict__ in, f16* __restrict__ out, int n) {
  int i = blockIdx.x * 256 + threadIdx.x;
  if (i < n) out[i] = (f16)in[i];
}

// ---------------- QKV projection GEMM ----------------
// Block = 8 waves, block tile 256x64; per k-step stage B(32x64) to LDS once.
// nb 0..7 -> Q cols, 8..15 -> K cols (stored transposed), 16..23 -> V cols.
__global__ void gemm_qkv_kernel(const f16* __restrict__ XN, const f16* __restrict__ Wqk,
                                const f16* __restrict__ Wv, f16* __restrict__ Q,
                                f16* __restrict__ Kt, f16* __restrict__ V) {
  __shared__ f16 bstage[32 * 64];
  int t = threadIdx.x;
  int lane = t & 31, wv = t >> 5, half = lane >> 4, l15 = lane & 15;
  int mb = blockIdx.x / 24, nb = blockIdx.x % 24;
  bool isqk = nb < 16;
  const f16* Bp = isqk ? Wqk : Wv;
  int ldb = isqk ? 1024 : 512;
  int n0  = isqk ? nb * 64 : (nb - 16) * 64;
  int m0 = mb * 256 + wv * 32;
  const f16* Arow0 = XN + (size_t)(m0 + l15) * DIMn;
  const f16* Arow1 = XN + (size_t)(m0 + 16 + l15) * DIMn;
  int sr = t >> 3, sc = (t & 7) * 8;

  v8f zero = {};
  v8f c[2][4];
#pragma unroll
  for (int r = 0; r < 2; ++r)
#pragma unroll
    for (int tt = 0; tt < 4; ++tt) c[r][tt] = zero;

  for (int k0 = 0; k0 < DIMn; k0 += 32) {
    stage16(Bp + (size_t)(k0 + sr) * ldb + n0 + sc, &bstage[sr * 64 + sc]);
    stage_wait();
    __syncthreads();
    int base = k0 + (half ? 8 : 0);
    v16h a0 = load_a_frag(Arow0, base);
    v16h a1 = load_a_frag(Arow1, base);
#pragma unroll
    for (int tt = 0; tt < 4; ++tt) {
      v16h b = load_b_frag(&bstage[tt * 16 + l15], 64, half);
      c[0][tt] = wmma_f16(a0, b, c[0][tt]);
      c[1][tt] = wmma_f16(a1, b, c[1][tt]);
    }
    __syncthreads();
  }

  int bidx = m0 >> 10;
  int nlo  = m0 & 1023;
  if (isqk && nb < 8) {              // ---- Q: [bh, n, d]
    int h = nb;
#pragma unroll
    for (int r = 0; r < 2; ++r)
#pragma unroll
      for (int tt = 0; tt < 4; ++tt)
#pragma unroll
        for (int j = 0; j < 8; ++j) {
          int n = nlo + r * 16 + j + 8 * half;
          Q[(((size_t)(bidx * Hn + h)) * Nn + n) * DHn + tt * 16 + l15] = (f16)c[r][tt][j];
        }
  } else if (isqk) {                 // ---- K: transposed [bh, d, n], packed b128 stores
    int h = nb - 8;
#pragma unroll
    for (int r = 0; r < 2; ++r)
#pragma unroll
      for (int tt = 0; tt < 4; ++tt) {
        PK pk;
#pragma unroll
        for (int j = 0; j < 8; ++j) pk.e[j] = (f16)c[r][tt][j];
        int d = tt * 16 + l15;
        int nbase = nlo + r * 16 + 8 * half;
        *(v4u*)&Kt[(((size_t)(bidx * Hn + h)) * DHn + d) * Nn + nbase] = pk.q;
      }
  } else {                           // ---- V: [bh, n, d]
    int h = nb - 16;
#pragma unroll
    for (int r = 0; r < 2; ++r)
#pragma unroll
      for (int tt = 0; tt < 4; ++tt)
#pragma unroll
        for (int j = 0; j < 8; ++j) {
          int n = nlo + r * 16 + j + 8 * half;
          V[(((size_t)(bidx * Hn + h)) * Nn + n) * DHn + tt * 16 + l15] = (f16)c[r][tt][j];
        }
  }
}

// ---------------- Flash attention ----------------
// Block = 8 waves, all sharing one (b,h); each wave owns 16 query rows.
// Per 32-key step: stage K^T(64x32) and V(32x64) chunks to LDS once per block.
__global__ void attn_kernel(const f16* __restrict__ Q, const f16* __restrict__ Kt,
                            const f16* __restrict__ V, f16* __restrict__ O) {
  __shared__ f16 kbuf[64 * 32];
  __shared__ f16 vbuf[32 * 64];
  __shared__ f16 plds[8][512];
  int t = threadIdx.x;
  int lane = t & 31, wv = t >> 5;
  int wid = blockIdx.x * 8 + wv;
  int bh = wid >> 6, qt = wid & 63;       // 8 blocks per bh -> bh uniform per block
  int half = lane >> 4, l15 = lane & 15;
  const f16* Qb  = Q  + ((size_t)bh * Nn + qt * 16) * DHn;
  const f16* Ktb = Kt + (size_t)bh * DHn * Nn;
  const f16* Vb  = V  + (size_t)bh * Nn * DHn;

  v16h q0 = load_a_frag(Qb + (size_t)l15 * DHn, 0  + (half ? 8 : 0));
  v16h q1 = load_a_frag(Qb + (size_t)l15 * DHn, 32 + (half ? 8 : 0));
  v8f o0 = {}, o1 = {}, o2 = {}, o3 = {};
  float m[8], l[8];
#pragma unroll
  for (int j = 0; j < 8; ++j) { m[j] = -1e30f; l[j] = 0.0f; }

  for (int it = 0; it < 32; ++it) {
    int n0 = it * 32;
    stage16(Ktb + (size_t)(t >> 2) * Nn + n0 + (t & 3) * 8, &kbuf[(t >> 2) * 32 + (t & 3) * 8]);
    stage16(Vb + (size_t)(n0 + (t >> 3)) * DHn + (t & 7) * 8, &vbuf[(t >> 3) * 64 + (t & 7) * 8]);
    stage_wait();
    __syncthreads();

    v8f S0 = {}, S1 = {};
    {
      v16h bk;
      bk = load_b_frag(&kbuf[0  * 32 + 0  + l15], 32, half); S0 = wmma_f16(q0, bk, S0);
      bk = load_b_frag(&kbuf[0  * 32 + 16 + l15], 32, half); S1 = wmma_f16(q0, bk, S1);
      bk = load_b_frag(&kbuf[32 * 32 + 0  + l15], 32, half); S0 = wmma_f16(q1, bk, S0);
      bk = load_b_frag(&kbuf[32 * 32 + 16 + l15], 32, half); S1 = wmma_f16(q1, bk, S1);
    }
    float fsc[8];
#pragma unroll
    for (int j = 0; j < 8; ++j) {
      float s0 = S0[j] * 0.125f, s1 = S1[j] * 0.125f;
      float rm = fmaxf(s0, s1);
      for (int mk = 1; mk < 16; mk <<= 1) rm = fmaxf(rm, __shfl_xor(rm, mk, 32));
      float mn = fmaxf(m[j], rm);
      float f  = __expf(m[j] - mn);
      float p0 = __expf(s0 - mn), p1 = __expf(s1 - mn);
      float rs = p0 + p1;
      for (int mk = 1; mk < 16; mk <<= 1) rs += __shfl_xor(rs, mk, 32);
      l[j] = l[j] * f + rs;
      m[j] = mn;
      fsc[j] = f;
      int row = j + 8 * half;
      plds[wv][row * 32 + l15]      = (f16)p0;
      plds[wv][row * 32 + l15 + 16] = (f16)p1;
    }
#pragma unroll
    for (int j = 0; j < 8; ++j) { o0[j] *= fsc[j]; o1[j] *= fsc[j]; o2[j] *= fsc[j]; o3[j] *= fsc[j]; }
    __syncthreads();
    v16h pA = load_a_frag(&plds[wv][l15 * 32], half ? 8 : 0);
    {
      v16h bv;
      bv = load_b_frag(&vbuf[0  + l15], 64, half); o0 = wmma_f16(pA, bv, o0);
      bv = load_b_frag(&vbuf[16 + l15], 64, half); o1 = wmma_f16(pA, bv, o1);
      bv = load_b_frag(&vbuf[32 + l15], 64, half); o2 = wmma_f16(pA, bv, o2);
      bv = load_b_frag(&vbuf[48 + l15], 64, half); o3 = wmma_f16(pA, bv, o3);
    }
    __syncthreads();
  }

  int b = bh >> 3, h = bh & 7;
#pragma unroll
  for (int j = 0; j < 8; ++j) {
    float inv = 1.0f / l[j];
    int n = qt * 16 + j + 8 * half;
    f16* op = O + ((size_t)b * Nn + n) * INNERn + h * DHn;
    op[l15]      = (f16)(o0[j] * inv);
    op[16 + l15] = (f16)(o1[j] * inv);
    op[32 + l15] = (f16)(o2[j] * inv);
    op[48 + l15] = (f16)(o3[j] * inv);
  }
}

// ---------------- Output projection: O16[8192,512] @ Wout16[512,512] -> f32 out ------
__global__ void gemm_out_kernel(const f16* __restrict__ O, const f16* __restrict__ W,
                                float* __restrict__ out) {
  __shared__ f16 bstage[32 * 64];
  int t = threadIdx.x;
  int lane = t & 31, wv = t >> 5, half = lane >> 4, l15 = lane & 15;
  int mb = blockIdx.x >> 3, nb = blockIdx.x & 7;
  int n0 = nb * 64;
  int m0 = mb * 256 + wv * 32;
  const f16* Arow0 = O + (size_t)(m0 + l15) * INNERn;
  const f16* Arow1 = O + (size_t)(m0 + 16 + l15) * INNERn;
  int sr = t >> 3, sc = (t & 7) * 8;

  v8f zero = {};
  v8f c[2][4];
#pragma unroll
  for (int r = 0; r < 2; ++r)
#pragma unroll
    for (int tt = 0; tt < 4; ++tt) c[r][tt] = zero;

  for (int k0 = 0; k0 < INNERn; k0 += 32) {
    stage16(W + (size_t)(k0 + sr) * DIMn + n0 + sc, &bstage[sr * 64 + sc]);
    stage_wait();
    __syncthreads();
    int base = k0 + (half ? 8 : 0);
    v16h a0 = load_a_frag(Arow0, base);
    v16h a1 = load_a_frag(Arow1, base);
#pragma unroll
    for (int tt = 0; tt < 4; ++tt) {
      v16h b = load_b_frag(&bstage[tt * 16 + l15], 64, half);
      c[0][tt] = wmma_f16(a0, b, c[0][tt]);
      c[1][tt] = wmma_f16(a1, b, c[1][tt]);
    }
    __syncthreads();
  }
#pragma unroll
  for (int r = 0; r < 2; ++r)
#pragma unroll
    for (int tt = 0; tt < 4; ++tt)
#pragma unroll
      for (int j = 0; j < 8; ++j)
        out[(size_t)(m0 + r * 16 + j + 8 * half) * DIMn + n0 + tt * 16 + l15] = c[r][tt][j];
}

extern "C" void kernel_launch(void* const* d_in, const int* in_sizes, int n_in,
                              void* d_out, int out_size, void* d_ws, size_t ws_size,
                              hipStream_t stream) {
  (void)in_sizes; (void)n_in; (void)out_size; (void)ws_size;
  const float* x     = (const float*)d_in[0];
  const float* gamma = (const float*)d_in[1];
  const float* beta  = (const float*)d_in[2];
  const float* Wqk   = (const float*)d_in[3];
  const float* Wv    = (const float*)d_in[4];
  const float* Wout  = (const float*)d_in[5];
  float* out = (float*)d_out;

  // workspace layout (f16 elements)
  f16* XN     = (f16*)d_ws;                       // 8192*512
  f16* Wqk16  = XN    + (size_t)8192 * 512;       // 512*1024
  f16* Wv16   = Wqk16 + (size_t)512 * 1024;       // 512*512
  f16* Wout16 = Wv16  + (size_t)512 * 512;        // 512*512
  f16* Qws    = Wout16+ (size_t)512 * 512;        // 64*1024*64
  f16* Ktws   = Qws   + (size_t)64 * 1024 * 64;   // 64*64*1024
  f16* Vws    = Ktws  + (size_t)64 * 64 * 1024;   // 64*1024*64
  f16* Ows    = Vws   + (size_t)64 * 1024 * 64;   // 8192*512

  ln_kernel<<<8192, 256, 0, stream>>>(x, gamma, beta, XN);
  cvt_kernel<<<(512 * 1024 + 255) / 256, 256, 0, stream>>>(Wqk, Wqk16, 512 * 1024);
  cvt_kernel<<<(512 * 512 + 255) / 256, 256, 0, stream>>>(Wv, Wv16, 512 * 512);
  cvt_kernel<<<(512 * 512 + 255) / 256, 256, 0, stream>>>(Wout, Wout16, 512 * 512);
  gemm_qkv_kernel<<<768, 256, 0, stream>>>(XN, Wqk16, Wv16, Qws, Ktws, Vws);
  attn_kernel<<<512, 256, 0, stream>>>(Qws, Ktws, Vws, Ows);
  gemm_out_kernel<<<256, 256, 0, stream>>>(Ows, Wout16, out);
}
